// VeloramaMLP_47768626266195
// MI455X (gfx1250) — matrix operand order repaired and checked by reading, not compiled, not run on hardware
//
#include <hip/hip_runtime.h>
#include <hip/hip_bf16.h>

// MI455X (gfx1250) fused Velorama MLP.
//
// GEMM0 (4096x2500 @ 2500x8000^T, 163.8 GFLOP) on v_wmma_f32_16x16x32_bf16.
// Each workgroup computes a 512-row x 4-target tile so every A fragment is
// reused by 4 B fragments (~33 FLOP/B from L2); packed bf16 operands (58.3 MB)
// are L2-resident (192 MB L2). Layers 1..3 (per-target 16x16 GEMM + dot) are
// fused in the epilogue so the 131 MB intermediate never touches HBM.

typedef __attribute__((ext_vector_type(16))) __bf16 v16bf;
typedef __attribute__((ext_vector_type(8)))  __bf16 v8bf;
typedef __attribute__((ext_vector_type(8)))  float  v8f;

#define LAG      5
#define NSAMP    4096
#define NREG     500
#define NT       500
#define HID      16
#define K_TRUE   (LAG * NREG)          // 2500
#define KPAD     2528                  // 79 * 32 (zero padded)
#define M_PER_WAVE 4                   // 4 x 16 = 64 rows per wave
#define NT_PER_BLOCK 4                 // 4 targets per workgroup
#define NWAVES   8
#define M_BLOCK  (NWAVES * M_PER_WAVE * 16)   // 512 rows per workgroup

static __device__ __forceinline__ unsigned short f32_to_bf16_rne(float f) {
    unsigned int u = __builtin_bit_cast(unsigned int, f);
    u += 0x7FFFu + ((u >> 16) & 1u);   // round to nearest even
    return (unsigned short)(u >> 16);
}

// ---- Pack AX [lag,N,nreg] f32 -> A [N][KPAD] bf16, k = l*NREG + r --------
__global__ void pack_A_kernel(const float* __restrict__ AX,
                              unsigned short* __restrict__ Ap) {
    long long idx = (long long)blockIdx.x * blockDim.x + threadIdx.x;
    const long long total = (long long)NSAMP * KPAD;
    if (idx >= total) return;
    int k = (int)(idx % KPAD);
    int n = (int)(idx / KPAD);
    unsigned short v = 0;
    if (k < K_TRUE) {
        int l = k / NREG, r = k % NREG;
        v = f32_to_bf16_rne(AX[((long long)l * NSAMP + n) * NREG + r]);
    }
    Ap[idx] = v;
}

// ---- Pack W0 [O,nreg,lag] f32 -> B [O][KPAD] bf16, B[o][l*NREG+r]=W0[o,r,4-l]
__global__ void pack_B_kernel(const float* __restrict__ W0,
                              unsigned short* __restrict__ Bp) {
    long long idx = (long long)blockIdx.x * blockDim.x + threadIdx.x;
    const long long total = (long long)(NT * HID) * KPAD;
    if (idx >= total) return;
    int k = (int)(idx % KPAD);
    int o = (int)(idx / KPAD);
    unsigned short v = 0;
    if (k < K_TRUE) {
        int l = k / NREG, r = k % NREG;
        v = f32_to_bf16_rne(W0[((long long)o * NREG + r) * LAG + (LAG - 1 - l)]);
    }
    Bp[idx] = v;
}

// ---- Fused: GEMM0 (WMMA bf16) + bias + relu + per-target 16x16 (WMMA) +
//      bias + relu + per-target dot + bias -> out[n, t] ---------------------
__global__ __launch_bounds__(256)
void velorama_fused_kernel(const unsigned short* __restrict__ Ap,
                           const unsigned short* __restrict__ Bp,
                           const float* __restrict__ b0,
                           const float* __restrict__ W1,
                           const float* __restrict__ b1,
                           const float* __restrict__ W2,
                           const float* __restrict__ b2,
                           float* __restrict__ out) {
    const int t0   = blockIdx.x * NT_PER_BLOCK;  // first target of this block
    const int rowB = blockIdx.y * M_BLOCK;       // sample-row base of block
    const int wave = threadIdx.x >> 5;           // wave32
    const int lane = threadIdx.x & 31;
    const int col  = lane & 15;                  // WMMA N index / M row-in-tile
    const int grp  = lane >> 4;                  // half-wave group

    __shared__ __bf16 lds_x[NWAVES][16][16];     // per-wave relu(x) tile

    // ------------------- big GEMM over K = KPAD -------------------
    // acc[m][j]: rows (wave,m) x 16 cols of target t0+j
    v8f acc[M_PER_WAVE][NT_PER_BLOCK];
    #pragma unroll
    for (int m = 0; m < M_PER_WAVE; ++m)
        #pragma unroll
        for (int j = 0; j < NT_PER_BLOCK; ++j) acc[m][j] = (v8f){};

    // Single per-lane base pointers; per-(m|j) deltas are compile-time
    // immediates that fold into the 24-bit instruction offsets.
    const unsigned short* bpBase = Bp + ((long long)(t0 * 16 + col)) * KPAD;
    const int waveRow0 = rowB + wave * (M_PER_WAVE * 16);
    const unsigned short* apBase = Ap + (long long)(waveRow0 + col) * KPAD;

    #pragma unroll 1   // keep accumulators in-place (no unroll ping-pong copies)
    for (int kc = 0; kc < KPAD; kc += 32) {
        // Speculative prefetch (safe past the end; fault is silently dropped).
        __builtin_prefetch(bpBase + kc + 32, 0, 1);

        // B fragments for the 4 targets: lane=col N, elems -> K=kc+grp*16+e
        v16bf bfrag[NT_PER_BLOCK];
        #pragma unroll
        for (int j = 0; j < NT_PER_BLOCK; ++j) {
            const unsigned short* bp = bpBase + (long long)j * 16 * KPAD + kc + grp * 16;
            v8bf blo = *(const v8bf*)(bp);
            v8bf bhi = *(const v8bf*)(bp + 8);
            #pragma unroll
            for (int i = 0; i < 8; ++i) { bfrag[j][i] = blo[i]; bfrag[j][8 + i] = bhi[i]; }
        }

        #pragma unroll
        for (int m = 0; m < M_PER_WAVE; ++m) {
            // A fragment: lane row=col, elems -> K=kc+grp*16+e   (2 x 16B contig)
            const unsigned short* ap = apBase + (long long)m * 16 * KPAD + kc;
            v8bf alo = *(const v8bf*)(ap + grp * 16);
            v8bf ahi = *(const v8bf*)(ap + grp * 16 + 8);
            v16bf afrag;
            #pragma unroll
            for (int i = 0; i < 8; ++i) { afrag[i] = alo[i]; afrag[8 + i] = ahi[i]; }

            #pragma unroll
            for (int j = 0; j < NT_PER_BLOCK; ++j) {
                acc[m][j] = __builtin_amdgcn_wmma_f32_16x16x32_bf16(
                    false, afrag, false, bfrag[j], (short)0, acc[m][j], false, false);
            }
        }
    }

    // ------------------- fused epilogue (layers 1..3) -------------------
    // Fully unrolled so acc[][] indices are compile-time constants: a rolled
    // loop would dynamically index the VGPR array and spill it to scratch.
    #pragma unroll
    for (int j = 0; j < NT_PER_BLOCK; ++j) {
        const int t = t0 + j;
        const float bias0 = b0[t * 16 + col];
        const float b1v   = b1[t * 16 + col];
        const float w2v   = W2[t * 16 + col];
        const float b2v   = b2[t];

        // Layer-1 B fragment: B[k=d][n=e] = W1[t,e,d]; K=16..31 zero padded.
        v16bf w1frag;
        #pragma unroll
        for (int i = 0; i < 16; ++i) w1frag[i] = (__bf16)0.0f;
        if (grp == 0) {
            const float* w1row = W1 + ((long long)t * 16 + col) * 16;
            #pragma unroll
            for (int i = 0; i < 16; ++i) w1frag[i] = (__bf16)w1row[i];
        }

        #pragma unroll
        for (int m = 0; m < M_PER_WAVE; ++m) {
            // relu(x + b0): C-layout (lane holds M = v + grp*8, N = col) -> LDS
            #pragma unroll
            for (int v = 0; v < 8; ++v) {
                float xv = acc[m][j][v] + bias0;
                xv = xv > 0.0f ? xv : 0.0f;
                lds_x[wave][v + grp * 8][col] = (__bf16)xv;
            }
            __syncthreads();

            // Layer-1 A fragment: row = col, grp 0 elems -> K = i, grp 1 all 0
            v16bf a2;
            #pragma unroll
            for (int i = 0; i < 16; ++i) a2[i] = (__bf16)0.0f;
            #pragma unroll
            for (int i = 0; i < 16; ++i) if (grp == 0) a2[i] = lds_x[wave][col][i];

            v8f y = (v8f){};
            y = __builtin_amdgcn_wmma_f32_16x16x32_bf16(
                false, a2, false, w1frag, (short)0, y, false, false);

            // Layer 2: relu(y + b1) . w2, reduce 16 lanes of each half-wave.
            #pragma unroll
            for (int v = 0; v < 8; ++v) {
                float yv = y[v] + b1v;
                yv = yv > 0.0f ? yv : 0.0f;
                float p = yv * w2v;
                p += __shfl_xor(p, 1, 32);
                p += __shfl_xor(p, 2, 32);
                p += __shfl_xor(p, 4, 32);
                p += __shfl_xor(p, 8, 32);
                if (col == 0) {
                    int row = waveRow0 + m * 16 + v + grp * 8;
                    out[(long long)row * NT + t] = p + b2v;
                }
            }
            __syncthreads();
        }
    }
}

extern "C" void kernel_launch(void* const* d_in, const int* in_sizes, int n_in,
                              void* d_out, int out_size, void* d_ws, size_t ws_size,
                              hipStream_t stream) {
    const float* AX = (const float*)d_in[0];   // [5,4096,500]
    const float* W0 = (const float*)d_in[1];   // [8000,500,5]
    const float* b0 = (const float*)d_in[2];   // [8000]
    const float* W1 = (const float*)d_in[3];   // [500,16,16]
    const float* b1 = (const float*)d_in[4];   // [500,16]
    const float* W2 = (const float*)d_in[5];   // [500,1,16]
    const float* b2 = (const float*)d_in[6];   // [500,1]
    float* out = (float*)d_out;                // [4096,500]

    // Workspace layout: A bf16 [4096][2528] then B bf16 [8000][2528] (~58.3 MB)
    unsigned short* Ap = (unsigned short*)d_ws;
    unsigned short* Bp = Ap + (size_t)NSAMP * KPAD;

    {
        long long tot = (long long)NSAMP * KPAD;
        pack_A_kernel<<<(unsigned)((tot + 255) / 256), 256, 0, stream>>>(AX, Ap);
    }
    {
        long long tot = (long long)(NT * HID) * KPAD;
        pack_B_kernel<<<(unsigned)((tot + 255) / 256), 256, 0, stream>>>(W0, Bp);
    }

    dim3 grid(NT / NT_PER_BLOCK, NSAMP / M_BLOCK);  // 125 x 8 workgroups
    velorama_fused_kernel<<<grid, 256, 0, stream>>>(Ap, Bp, b0, W1, b1, W2, b2, out);
}
